// GraphFormer_positional_embedding_44341242364111
// MI455X (gfx1250) — matrix operand order repaired and checked by reading, not compiled, not run on hardware
//
#include <hip/hip_runtime.h>
#include <cstdint>
#include <cstddef>

// Problem constants (match reference)
#define B_  4
#define N_  2048
#define DD  128
#define KK  32
#define EPS 1e-5f

typedef __attribute__((ext_vector_type(16))) __bf16 v16bf;
typedef __attribute__((ext_vector_type(8)))  float  v8f;

// ---------------------------------------------------------------------------
// float -> bf16 with round-to-nearest-even (bit manipulation, no fp conv dep)
// ---------------------------------------------------------------------------
__device__ __forceinline__ __bf16 f2bf(float f) {
  unsigned u = __builtin_bit_cast(unsigned, f);
  unsigned r = u + 0x7FFFu + ((u >> 16) & 1u);
  return __builtin_bit_cast(__bf16, (unsigned short)(r >> 16));
}

// ---------------------------------------------------------------------------
// CDNA5 async global->LDS copy (ASYNCcnt-tracked). Per-lane 16B transfer.
// Generic LDS pointer: low 32 bits are the LDS byte offset (ISA 10.2 aperture
// rules), so truncation yields the DS address for the VDST operand.
// ---------------------------------------------------------------------------
__device__ __forceinline__ void async_load_b128(const void* gaddr, void* ldsaddr) {
  unsigned lds = (unsigned)(size_t)ldsaddr;
  asm volatile("global_load_async_to_lds_b128 %0, %1, off"
               :: "v"(lds), "v"(gaddr) : "memory");
}
__device__ __forceinline__ void wait_async() {
  asm volatile("s_wait_asynccnt 0x0" ::: "memory");
}

// ---------------------------------------------------------------------------
// WMMA 16-bit A-matrix lane layout (CDNA5 ISA 7.12.2, 16x32 MxK):
//   lanes 0-15 : VGPR0-3 K=0..7,  VGPR4-7 K=16..23   (M = lane)
//   lanes 16-31: VGPR0-3 K=8..15, VGPR4-7 K=24..31   (M = lane-16)
// apack buffer layout: [mtile(2)][ktile(4)][lane(32)][elem(16)] bf16
// ---------------------------------------------------------------------------
__device__ __forceinline__ int apackOff(int m, int k) {
  int mt = m >> 4, kt = k >> 5, kk = k & 31;
  int lane = (m & 15) + (((kk >> 3) & 1) << 4);
  int e = (kk & 7) + ((kk >> 4) << 3);
  return ((mt * 4 + kt) * 32 + lane) * 16 + e;
}

// ---------------------------------------------------------------------------
// One [32,128] = [32,128]x[128,128] matmul on a 256-thread block (8 waves).
// Wave w computes output column tile n in [16w,16w+16) for both row tiles.
// Wp layout: [ktile(4)][ntile(8)][lane(32)][elem(16)] bf16 (B-matrix packed:
//   lane L holds column n = L%16; K = (e>>3)*16 + (L<16?0:8) + (e&7) within the 32-K tile).
// ---------------------------------------------------------------------------
__device__ __forceinline__ void wmma_tile32(const __bf16* sA, const __bf16* Wp,
                                            const float* bias, float* sOut,
                                            int wave, int lane, bool relu) {
  v8f c0 = {};
  v8f c1 = {};
#pragma unroll
  for (int kt = 0; kt < 4; ++kt) {
    v16bf a0 = *(const v16bf*)(sA + ((0 * 4 + kt) * 32 + lane) * 16);
    v16bf a1 = *(const v16bf*)(sA + ((1 * 4 + kt) * 32 + lane) * 16);
    v16bf b  = *(const v16bf*)(Wp + (((kt * 8 + wave) * 32) + lane) * 16);
    c0 = __builtin_amdgcn_wmma_f32_16x16x32_bf16(false, a0, false, b, (short)0, c0, false, false);
    c1 = __builtin_amdgcn_wmma_f32_16x16x32_bf16(false, a1, false, b, (short)0, c1, false, false);
  }
  int n = wave * 16 + (lane & 15);
  float bv = bias[n];
  int mb = (lane < 16) ? 0 : 8;
#pragma unroll
  for (int r = 0; r < 8; ++r) {
    float v0 = c0[r] + bv, v1 = c1[r] + bv;
    if (relu) { v0 = fmaxf(v0, 0.f); v1 = fmaxf(v1, 0.f); }
    sOut[(mb + r) * DD + n]        = v0;   // C/D layout: VGPR r -> M = r(+8), N = lane%16
    sOut[(16 + mb + r) * DD + n]   = v1;
  }
}

// ---------------------------------------------------------------------------
// Pack a [128,128] fp32 weight (fan_in x fan_out, row major) into the
// WMMA B-matrix lane layout, bf16.
// ---------------------------------------------------------------------------
__global__ void packw_kernel(const float* __restrict__ W, __bf16* __restrict__ Wp) {
  int t = blockIdx.x * blockDim.x + threadIdx.x;       // 0..16383
  int e  = t & 15;
  int L  = (t >> 4) & 31;
  int nt = (t >> 9) & 7;
  int kt = t >> 12;
  int k = kt * 32 + ((e >> 3) << 4) + ((L >= 16) ? 8 : 0) + (e & 7);
  int n = nt * 16 + (L & 15);
  Wp[t] = f2bf(W[k * DD + n]);
}

// ---------------------------------------------------------------------------
// Brute-force KNN: one wave per query. Selects the 33 smallest d2; entry 0 is
// self (d2=0) -> idx = first 32 (self included), idx2 = entries 1..32.
// ---------------------------------------------------------------------------
__global__ void knn_kernel(const float* __restrict__ coords,
                           int* __restrict__ idx, int* __restrict__ idx2) {
  int wave = threadIdx.x >> 5, lane = threadIdx.x & 31;
  int q = blockIdx.x * 8 + wave;          // global query in [0, B*N)
  int b = q / N_, i = q - b * N_;
  const float* cb = coords + (size_t)b * N_ * 3;
  float cx = cb[i * 3 + 0], cy = cb[i * 3 + 1], cz = cb[i * 3 + 2];

  float dist[64];
  for (int s = 0; s < 64; ++s) {
    int j = lane + 32 * s;
    float dx = cb[j * 3 + 0] - cx;
    float dy = cb[j * 3 + 1] - cy;
    float dz = cb[j * 3 + 2] - cz;
    dist[s] = dx * dx + dy * dy + dz * dz;
  }
  const float INF = 3.0e38f;
  for (int sel = 0; sel < 33; ++sel) {
    float best = INF; int bs = 0;
    for (int s = 0; s < 64; ++s)
      if (dist[s] < best) { best = dist[s]; bs = s; }
    float rb = best; int rl = lane;
    for (int off = 16; off; off >>= 1) {
      float ob = __shfl_xor(rb, off, 32);
      int   ol = __shfl_xor(rl, off, 32);
      if (ob < rb || (ob == rb && ol < rl)) { rb = ob; rl = ol; }
    }
    int wbs = __shfl(bs, rl, 32);
    int j = rl + 32 * wbs;
    if (lane == rl) dist[wbs] = INF;
    if (lane == 0) {
      if (sel < 32) idx[(size_t)q * KK + sel] = j;
      if (sel >= 1) idx2[(size_t)q * KK + sel - 1] = j;
    }
  }
}

// ---------------------------------------------------------------------------
// Attention kernel: one 256-thread block per query point.
// fc -> pe -> {V, Kf} -> rel -> w (2 layers) -> sparsemax -> x_att
// All five [32,128]x[128,128] matmuls via WMMA bf16. Q precomputed (WMMA).
// Neighbor feature rows gathered with async global->LDS copies, overlapped
// with the fc matmul. sVX doubles as gather target then V storage.
// ---------------------------------------------------------------------------
__global__ void attn_kernel(const float* __restrict__ feats,
                            const float* __restrict__ coords,
                            const int* __restrict__ idx,
                            const float* __restrict__ Qg,
                            const float* __restrict__ kl_b, const float* __restrict__ vl_b,
                            const float* __restrict__ pl1_w, const float* __restrict__ pl1_b,
                            const float* __restrict__ pl2_b,
                            const float* __restrict__ we1_b, const float* __restrict__ we2_b,
                            const __bf16* __restrict__ Wpl2, const __bf16* __restrict__ Wkl,
                            const __bf16* __restrict__ Wvl, const __bf16* __restrict__ Wwe1,
                            const __bf16* __restrict__ Wwe2,
                            float* __restrict__ x_att) {
  __shared__ alignas(64) __bf16 sA[2 * 4 * 32 * 16];   // WMMA A pack (8 KB)
  __shared__ float sOut[32 * DD];                      // matmul output (16 KB)
  __shared__ float sFc[32 * DD];                       // fc, kept fp32 (16 KB)
  __shared__ alignas(16) float sVX[32 * DD];           // gather target, then V (16 KB)
  __shared__ float sQ[DD];
  __shared__ float sFeat[DD];
  __shared__ float sNC[KK * 3];                        // neighbor coord diffs
  __shared__ int   sIdx[KK];

  int tid = threadIdx.x, wave = tid >> 5, lane = tid & 31;
  int q = blockIdx.x;
  int b = q / N_, i = q - b * N_;
  const float* fb = feats + (size_t)b * N_ * DD;
  const float* cb = coords + (size_t)b * N_ * 3;

  if (tid < KK) sIdx[tid] = idx[(size_t)q * KK + tid];
  if (tid >= 128 && tid < 256) sFeat[tid - 128] = fb[(size_t)i * DD + tid - 128];
  if (tid < DD) sQ[tid] = Qg[(size_t)q * DD + tid];
  __syncthreads();

  // async gather of 32 neighbor feature rows -> sVX (overlaps fc matmul)
#pragma unroll
  for (int r = 0; r < 4; ++r) {
    int f4 = r * 256 + tid;            // 1024 float4s total
    int m = f4 >> 5;                   // 32 float4 per row
    int c = (f4 & 31) * 4;
    async_load_b128(fb + (size_t)sIdx[m] * DD + c, sVX + m * DD + c);
  }
  if (tid < KK * 3) {
    int m = tid / 3, j = tid - m * 3;
    sNC[tid] = cb[(size_t)sIdx[m] * 3 + j] - cb[(size_t)i * 3 + j];
  }
  __syncthreads();

  // fc hidden = relu(diff @ pl1 + b)  ([32,3]x[3,128], VALU) -> packed A
#pragma unroll
  for (int e = 0; e < 16; ++e) {
    int elem = tid * 16 + e; int m = elem >> 7; int d = elem & 127;
    float acc = pl1_b[d]
              + sNC[m * 3 + 0] * pl1_w[0 * DD + d]
              + sNC[m * 3 + 1] * pl1_w[1 * DD + d]
              + sNC[m * 3 + 2] * pl1_w[2 * DD + d];
    sA[apackOff(m, d)] = f2bf(fmaxf(acc, 0.f));
  }
  __syncthreads();
  wmma_tile32(sA, Wpl2, pl2_b, sFc, wave, lane, false);       // fc
  wait_async();                                               // gather landed
  __syncthreads();

  // pe = fc + feat[idx]  -> packed A
#pragma unroll
  for (int e = 0; e < 16; ++e) {
    int elem = tid * 16 + e; int m = elem >> 7; int d = elem & 127;
    float pe = sFc[m * DD + d] + sVX[m * DD + d];
    sA[apackOff(m, d)] = f2bf(pe);
  }
  __syncthreads();
  wmma_tile32(sA, Wvl, vl_b, sVX, wave, lane, false);         // V (overwrites gather)
  wmma_tile32(sA, Wkl, kl_b, sOut, wave, lane, true);         // Kf = relu(...)
  __syncthreads();

  // rel = (Kf - Q) * fc -> packed A
#pragma unroll
  for (int e = 0; e < 16; ++e) {
    int elem = tid * 16 + e; int m = elem >> 7; int d = elem & 127;
    float rel = (sOut[m * DD + d] - sQ[d]) * sFc[m * DD + d];
    sA[apackOff(m, d)] = f2bf(rel);
  }
  __syncthreads();
  wmma_tile32(sA, Wwe1, we1_b, sOut, wave, lane, true);       // relu hidden
  __syncthreads();
#pragma unroll
  for (int e = 0; e < 16; ++e) {
    int elem = tid * 16 + e; int m = elem >> 7; int d = elem & 127;
    sA[apackOff(m, d)] = f2bf(sOut[m * DD + d]);
  }
  __syncthreads();
  wmma_tile32(sA, Wwe2, we2_b, sOut, wave, lane, false);      // w scores
  __syncthreads();

  // sparsemax over the K axis, one channel per thread (threads 0..127)
  if (tid < DD) {
    float z[KK], zs[KK];
    for (int k = 0; k < KK; ++k) { z[k] = sOut[k * DD + tid]; zs[k] = z[k]; }
    for (int a = 1; a < KK; ++a) {            // insertion sort, descending
      float key = zs[a]; int b2 = a - 1;
      while (b2 >= 0 && zs[b2] < key) { zs[b2 + 1] = zs[b2]; --b2; }
      zs[b2 + 1] = key;
    }
    float csum = 0.f, tau = 0.f;
    for (int r = 0; r < KK; ++r) {
      csum += zs[r];
      if ((float)(r + 1) * zs[r] > csum - 1.f) tau = (csum - 1.f) / (float)(r + 1);
    }
    for (int k = 0; k < KK; ++k) sOut[k * DD + tid] = fmaxf(z[k] - tau, 0.f);
  }
  __syncthreads();

  // x_att = feat + sum_k V*w
  if (tid < DD) {
    float acc = sFeat[tid];
    for (int k = 0; k < KK; ++k) acc += sVX[k * DD + tid] * sOut[k * DD + tid];
    x_att[(size_t)q * DD + tid] = acc;
  }
}

// ---------------------------------------------------------------------------
// Generic [32,128]x[128,128] WMMA GEMM over rows of X: Y = act(X @ W + b)
// X tile staged into LDS with async global->LDS copies.
// ---------------------------------------------------------------------------
__global__ void gemm32_kernel(const float* __restrict__ X, const __bf16* __restrict__ Wp,
                              const float* __restrict__ bias, float* __restrict__ Y,
                              int relu) {
  __shared__ alignas(64) __bf16 sA[2 * 4 * 32 * 16];
  __shared__ alignas(16) float sX[32 * DD];
  __shared__ float sOut[32 * DD];
  int tid = threadIdx.x, wave = tid >> 5, lane = tid & 31;
  size_t row0 = (size_t)blockIdx.x * 32;
  __builtin_prefetch(Wp + (size_t)tid * 64, 0, 1);    // global_prefetch_b8
#pragma unroll
  for (int r = 0; r < 4; ++r) {
    int f4 = r * 256 + tid;                           // 1024 float4s = 32x128
    async_load_b128(X + row0 * DD + (size_t)f4 * 4, sX + f4 * 4);
  }
  wait_async();
  __syncthreads();
#pragma unroll
  for (int e = 0; e < 16; ++e) {
    int elem = tid * 16 + e; int m = elem >> 7; int d = elem & 127;
    sA[apackOff(m, d)] = f2bf(sX[m * DD + d]);
  }
  __syncthreads();
  wmma_tile32(sA, Wp, bias, sOut, wave, lane, relu != 0);
  __syncthreads();
#pragma unroll
  for (int e = 0; e < 16; ++e) {
    int elem = tid * 16 + e; int m = elem >> 7; int d = elem & 127;
    Y[(row0 + m) * DD + d] = sOut[m * DD + d];
  }
}

// cf hidden = relu(coord @ pe1 + b)    ([B*N,3] x [3,128])
__global__ void pe_hidden_kernel(const float* __restrict__ coords,
                                 const float* __restrict__ pe1_w,
                                 const float* __restrict__ pe1_b,
                                 float* __restrict__ out) {
  int gid = blockIdx.x * blockDim.x + threadIdx.x;
  int d = gid & 127; size_t row = (size_t)gid >> 7;
  const float* c = coords + row * 3;
  float acc = pe1_b[d] + c[0] * pe1_w[d] + c[1] * pe1_w[DD + d] + c[2] * pe1_w[2 * DD + d];
  out[gid] = fmaxf(acc, 0.f);
}

// y = a + b
__global__ void add_kernel(const float* __restrict__ a, const float* __restrict__ b,
                           float* __restrict__ y) {
  int gid = blockIdx.x * blockDim.x + threadIdx.x;
  y[gid] = a[gid] + b[gid];
}

// LayerNorm(x [+ add]) per row of 128, one wave per row
__global__ void ln_kernel(const float* __restrict__ x, const float* __restrict__ add,
                          const float* __restrict__ g, const float* __restrict__ bt,
                          float* __restrict__ y) {
  int wave = threadIdx.x >> 5, lane = threadIdx.x & 31;
  size_t row = (size_t)blockIdx.x * 8 + wave;
  float v[4];
#pragma unroll
  for (int e = 0; e < 4; ++e) {
    int d = lane + 32 * e;
    v[e] = x[row * DD + d] + (add ? add[row * DD + d] : 0.f);
  }
  float s = v[0] + v[1] + v[2] + v[3];
  for (int off = 16; off; off >>= 1) s += __shfl_xor(s, off, 32);
  float mean = s * (1.f / 128.f);
  float sq = 0.f;
#pragma unroll
  for (int e = 0; e < 4; ++e) { float t = v[e] - mean; sq += t * t; }
  for (int off = 16; off; off >>= 1) sq += __shfl_xor(sq, off, 32);
  float inv = rsqrtf(sq * (1.f / 128.f) + EPS);
#pragma unroll
  for (int e = 0; e < 4; ++e) {
    int d = lane + 32 * e;
    y[row * DD + d] = (v[e] - mean) * inv * g[d] + bt[d];
  }
}

// SGConv propagation: deg == K+1 == 33 for every node (K in-edges + self loop),
// so D^{-1/2}(A+I)D^{-1/2} x collapses to (x[i] + sum_k x[nbr_k]) / 33.
__global__ void prop_kernel(const float* __restrict__ x, const int* __restrict__ idx2,
                            float* __restrict__ y) {
  __shared__ int sIdx[KK];
  size_t q = blockIdx.x; int d = threadIdx.x;
  int b = (int)(q / N_);
  size_t base = (size_t)b * N_ * DD;
  if (d < KK) sIdx[d] = idx2[q * KK + d];
  __syncthreads();
  float acc = x[q * DD + d];
  for (int k = 0; k < KK; ++k) acc += x[base + (size_t)sIdx[k] * DD + d];
  y[q * DD + d] = acc * (1.f / 33.f);
}

// ---------------------------------------------------------------------------
extern "C" void kernel_launch(void* const* d_in, const int* in_sizes, int n_in,
                              void* d_out, int out_size, void* d_ws, size_t ws_size,
                              hipStream_t stream) {
  const float* features = (const float*)d_in[0];
  const float* coords   = (const float*)d_in[1];
  const float* ql_w  = (const float*)d_in[2];  const float* ql_b  = (const float*)d_in[3];
  const float* kl_w  = (const float*)d_in[4];  const float* kl_b  = (const float*)d_in[5];
  const float* vl_w  = (const float*)d_in[6];  const float* vl_b  = (const float*)d_in[7];
  const float* pl1_w = (const float*)d_in[8];  const float* pl1_b = (const float*)d_in[9];
  const float* pl2_w = (const float*)d_in[10]; const float* pl2_b = (const float*)d_in[11];
  const float* we1_w = (const float*)d_in[12]; const float* we1_b = (const float*)d_in[13];
  const float* we2_w = (const float*)d_in[14]; const float* we2_b = (const float*)d_in[15];
  const float* pe1_w = (const float*)d_in[16]; const float* pe1_b = (const float*)d_in[17];
  const float* pe2_w = (const float*)d_in[18]; const float* pe2_b = (const float*)d_in[19];
  const float* sg1_w = (const float*)d_in[20]; const float* sg1_b = (const float*)d_in[21];
  const float* sg2_w = (const float*)d_in[22]; const float* sg2_b = (const float*)d_in[23];
  const float* lnA_g = (const float*)d_in[24]; const float* lnA_b = (const float*)d_in[25];
  const float* lnF_g = (const float*)d_in[26]; const float* lnF_b = (const float*)d_in[27];
  (void)in_sizes; (void)n_in; (void)out_size; (void)ws_size;

  // ---- carve workspace ----
  char* w = (char*)d_ws;
  auto carve = [&](size_t bytes) -> char* {
    char* r = w; w += (bytes + 255) & ~(size_t)255; return r;
  };
  const size_t WPACK = 16384 * sizeof(__bf16);
  __bf16* Wql  = (__bf16*)carve(WPACK);
  __bf16* Wpl2 = (__bf16*)carve(WPACK);
  __bf16* Wkl  = (__bf16*)carve(WPACK);
  __bf16* Wvl  = (__bf16*)carve(WPACK);
  __bf16* Wwe1 = (__bf16*)carve(WPACK);
  __bf16* Wwe2 = (__bf16*)carve(WPACK);
  __bf16* Wpe2 = (__bf16*)carve(WPACK);
  __bf16* Wsg1 = (__bf16*)carve(WPACK);
  __bf16* Wsg2 = (__bf16*)carve(WPACK);
  int* idx  = (int*)carve((size_t)B_ * N_ * KK * sizeof(int));
  int* idx2 = (int*)carve((size_t)B_ * N_ * KK * sizeof(int));
  const size_t FB = (size_t)B_ * N_ * DD * sizeof(float);
  float* buf0 = (float*)carve(FB);
  float* buf1 = (float*)carve(FB);
  float* buf2 = (float*)carve(FB);
  float* buf3 = (float*)carve(FB);
  float* out = (float*)d_out;

  // ---- pack weights to WMMA B-matrix bf16 layout ----
  packw_kernel<<<64, 256, 0, stream>>>(ql_w,  Wql);
  packw_kernel<<<64, 256, 0, stream>>>(pl2_w, Wpl2);
  packw_kernel<<<64, 256, 0, stream>>>(kl_w,  Wkl);
  packw_kernel<<<64, 256, 0, stream>>>(vl_w,  Wvl);
  packw_kernel<<<64, 256, 0, stream>>>(we1_w, Wwe1);
  packw_kernel<<<64, 256, 0, stream>>>(we2_w, Wwe2);
  packw_kernel<<<64, 256, 0, stream>>>(pe2_w, Wpe2);
  packw_kernel<<<64, 256, 0, stream>>>(sg1_w, Wsg1);
  packw_kernel<<<64, 256, 0, stream>>>(sg2_w, Wsg2);

  // ---- KNN (idx with self, idx2 without) ----
  knn_kernel<<<B_ * N_ / 8, 256, 0, stream>>>(coords, idx, idx2);

  // ---- Q = relu(feat @ ql + b) for all rows (WMMA) -> buf1 ----
  gemm32_kernel<<<B_ * N_ / 32, 256, 0, stream>>>(features, Wql, ql_b, buf1, 1);

  // ---- attention block -> x_att (buf0) ----
  attn_kernel<<<B_ * N_, 256, 0, stream>>>(features, coords, idx, buf1,
                                           kl_b, vl_b, pl1_w, pl1_b, pl2_b,
                                           we1_b, we2_b,
                                           Wpl2, Wkl, Wvl, Wwe1, Wwe2, buf0);

  // ---- x_in = LN_A(x_att) (buf2) ----
  ln_kernel<<<B_ * N_ / 8, 256, 0, stream>>>(buf0, nullptr, lnA_g, lnA_b, buf2);

  // ---- cf = relu(coord@pe1+b)@pe2+b ; x_pe = x_in + cf ----
  pe_hidden_kernel<<<B_ * N_ * DD / 256, 256, 0, stream>>>(coords, pe1_w, pe1_b, buf1);
  gemm32_kernel<<<B_ * N_ / 32, 256, 0, stream>>>(buf1, Wpe2, pe2_b, buf3, 0);     // cf
  add_kernel<<<B_ * N_ * DD / 256, 256, 0, stream>>>(buf2, buf3, buf1);            // x_pe

  // ---- SGConv x2 ----
  prop_kernel<<<B_ * N_, DD, 0, stream>>>(buf1, idx2, buf0);                       // p1
  gemm32_kernel<<<B_ * N_ / 32, 256, 0, stream>>>(buf0, Wsg1, sg1_b, buf3, 1);     // h1=relu
  prop_kernel<<<B_ * N_, DD, 0, stream>>>(buf3, idx2, buf0);                       // p2
  gemm32_kernel<<<B_ * N_ / 32, 256, 0, stream>>>(buf0, Wsg2, sg2_b, buf3, 0);     // h2

  // ---- out = LN_F(h2 + x_in) ----
  ln_kernel<<<B_ * N_ / 8, 256, 0, stream>>>(buf3, buf2, lnF_g, lnF_b, out);
}
